// AttentionGCNLayer_18769007083644
// MI455X (gfx1250) — matrix operand order repaired and checked by reading, not compiled, not run on hardware
//
#include <hip/hip_runtime.h>

// Problem constants (match reference)
#define B_     8
#define N_     1024
#define D_     768
#define NHEAD_ 4
#define DK_    192
#define NE_    3
#define ITERS_ 2
#define M_TOT  (B_ * N_)          // 8192 rows for (b,n)-flattened matrices

typedef __bf16 bf16_t;
typedef __attribute__((ext_vector_type(8)))  bf16_t v8bf;
typedef __attribute__((ext_vector_type(16))) bf16_t v16bf;
typedef __attribute__((ext_vector_type(8)))  float  v8f;

static __device__ __forceinline__ bf16_t f2bf(float f) {
    unsigned u = __builtin_bit_cast(unsigned, f);
    unsigned r = (u + 0x7FFFu + ((u >> 16) & 1u)) >> 16;   // round-to-nearest-even
    return __builtin_bit_cast(bf16_t, (unsigned short)r);
}

static __device__ __forceinline__ v8f vzero8() {
    v8f z;
#pragma unroll
    for (int i = 0; i < 8; i++) z[i] = 0.0f;
    return z;
}

// A fragment (16x32 bf16, row-major source).
// lane m = lane&15, half = lane>>4. Per ISA layout:
//   lanes 0-15 : K = {kt+0..7, kt+16..23};  lanes 16-31 : K = {kt+8..15, kt+24..31}
static __device__ __forceinline__ v16bf load_frag_a(const bf16_t* row, int k, int hf) {
    v8bf lo = *(const v8bf*)(row + k + 8 * hf);
    v8bf hi = *(const v8bf*)(row + k + 16 + 8 * hf);
    return __builtin_shufflevector(lo, hi, 0,1,2,3,4,5,6,7,8,9,10,11,12,13,14,15);
}

// B fragment (32x16 bf16) from TRANSPOSED B (Bt is Ncols x K row-major).
// lane n = lane&15 holds column n: lanes 0-15 K={kt+0..7, kt+16..23}, lanes 16-31 K={kt+8..15, kt+24..31}
static __device__ __forceinline__ v16bf load_frag_b(const bf16_t* row, int k, int hf) {
    const bf16_t* p = row + k + 8 * hf;
    v8bf lo = *(const v8bf*)(p);
    v8bf hi = *(const v8bf*)(p + 16);
    return __builtin_shufflevector(lo, hi, 0,1,2,3,4,5,6,7,8,9,10,11,12,13,14,15);
}

#define WMMA_BF16(a, b, c) \
    __builtin_amdgcn_wmma_f32_16x16x32_bf16(false, (a), false, (b), (short)0, (c), false, false)

// ---------------------------------------------------------------- utilities

__global__ void cvt_f32_to_bf16(const float* __restrict__ in, bf16_t* __restrict__ out, long n) {
    for (long i = (long)blockIdx.x * blockDim.x + threadIdx.x; i < n;
         i += (long)gridDim.x * blockDim.x)
        out[i] = f2bf(in[i]);
}

// out[c*R + r] = bf16(in[r*C + c]); batched via blockIdx.z
__global__ void transpose_to_bf16(const float* __restrict__ in, bf16_t* __restrict__ out,
                                  int R, int C) {
    long total = (long)R * C;
    const float* src = in + (long)blockIdx.z * total;
    bf16_t* dst = out + (long)blockIdx.z * total;
    for (long i = (long)blockIdx.x * blockDim.x + threadIdx.x; i < total;
         i += (long)gridDim.x * blockDim.x) {
        long r = i / C, c = i % C;
        dst[c * (long)R + r] = f2bf(src[i]);
    }
}

// X[h][b][n][d] = nodes[b][n][d] (f32 + bf16 copies)
__global__ void init_X(const float* __restrict__ nodes, float* __restrict__ Xf,
                       bf16_t* __restrict__ Xbf, long per, long total) {
    for (long i = (long)blockIdx.x * blockDim.x + threadIdx.x; i < total;
         i += (long)gridDim.x * blockDim.x) {
        float v = nodes[i % per];
        Xf[i] = v;
        Xbf[i] = f2bf(v);
    }
}

// ---------------------------------------------------------------- projection GEMM
// Out(bf16, MxNcols) = A(bf16, MxK) * Bt^T + bias ; wave tile = 16 x 64
// Double-buffered fragments: loads for k+32 issued before WMMAs of k.
__global__ void gemm_proj(const bf16_t* __restrict__ A, const bf16_t* __restrict__ Bt,
                          const float* __restrict__ bias, bf16_t* __restrict__ Out,
                          int M, int Ncols, int K) {
    int wave = blockIdx.x * (blockDim.x >> 5) + (threadIdx.x >> 5);
    int mtc = M >> 4, ngc = Ncols >> 6;
    if (wave >= mtc * ngc) return;
    int mt = wave % mtc, ng = wave / mtc;
    int lane = threadIdx.x & 31, l16 = lane & 15, hf = lane >> 4;

    const bf16_t* Arow  = A  + (long)(mt * 16 + l16) * K;
    const bf16_t* Brow0 = Bt + (long)(ng * 64 + l16) * K;

    v8f acc[4];
#pragma unroll
    for (int j = 0; j < 4; j++) acc[j] = vzero8();

    v16bf a0 = load_frag_a(Arow, 0, hf);
    v16bf b0[4];
#pragma unroll
    for (int j = 0; j < 4; j++) b0[j] = load_frag_b(Brow0 + (long)(16 * j) * K, 0, hf);

    for (int k = 32; k < K; k += 32) {
        __builtin_prefetch(Arow + k + 128, 0, 3);
        v16bf a1 = load_frag_a(Arow, k, hf);
        v16bf b1[4];
#pragma unroll
        for (int j = 0; j < 4; j++) b1[j] = load_frag_b(Brow0 + (long)(16 * j) * K, k, hf);
#pragma unroll
        for (int j = 0; j < 4; j++) acc[j] = WMMA_BF16(a0, b0[j], acc[j]);
        a0 = a1;
#pragma unroll
        for (int j = 0; j < 4; j++) b0[j] = b1[j];
    }
#pragma unroll
    for (int j = 0; j < 4; j++) acc[j] = WMMA_BF16(a0, b0[j], acc[j]);

#pragma unroll
    for (int j = 0; j < 4; j++) {
        int n = ng * 64 + 16 * j + l16;
        float bs = bias[n];
#pragma unroll
        for (int r = 0; r < 8; r++) {
            int m = mt * 16 + r + 8 * hf;
            Out[(long)m * Ncols + n] = f2bf(acc[j][r] + bs);
        }
    }
}

// ---------------------------------------------------------------- edge-masked scores
// ga[(h*B+b)][n][m] = sum_e (adj[b][n][m]==e+1 ? (Q_e K_e^T)*scale : 0); 0 -> -1e9
__global__ void scores_kernel(const bf16_t* __restrict__ Qbf, const bf16_t* __restrict__ Kbf,
                              const int* __restrict__ adj, float* __restrict__ ga) {
    int slice = blockIdx.y;               // h*B + b
    int h = slice >> 3, b = slice & (B_ - 1);
    int wave = blockIdx.x * (blockDim.x >> 5) + (threadIdx.x >> 5);
    if (wave >= 64 * 64) return;
    int mt = wave & 63, nt = wave >> 6;
    int lane = threadIdx.x & 31, l16 = lane & 15, hf = lane >> 4;
    int m0 = mt * 16, n0 = nt * 16;

    const int* adjp = adj + ((long)b * N_ + m0) * N_ + n0 + l16;
    int adjv[8];
#pragma unroll
    for (int r = 0; r < 8; r++) adjv[r] = adjp[(long)(r + 8 * hf) * N_];

    long qoff = (long)b * N_ * D_ + (long)h * DK_;
    float accsel[8];
#pragma unroll
    for (int r = 0; r < 8; r++) accsel[r] = 0.0f;
    const float scale = 0.07216878364870322f;   // 1/sqrt(192)

    for (int e = 0; e < NE_; e++) {
        const bf16_t* Arow = Qbf + (long)e * M_TOT * D_ + qoff + (long)(m0 + l16) * D_;
        const bf16_t* Brow = Kbf + (long)e * M_TOT * D_ + qoff + (long)(n0 + l16) * D_;
        v8f acc = vzero8();
        v16bf a0 = load_frag_a(Arow, 0, hf);
        v16bf b0 = load_frag_b(Brow, 0, hf);
#pragma unroll
        for (int k = 32; k < DK_; k += 32) {
            v16bf a1 = load_frag_a(Arow, k, hf);
            v16bf b1 = load_frag_b(Brow, k, hf);
            acc = WMMA_BF16(a0, b0, acc);
            a0 = a1; b0 = b1;
        }
        acc = WMMA_BF16(a0, b0, acc);
        int match = e + 1;
#pragma unroll
        for (int r = 0; r < 8; r++)
            if (adjv[r] == match) accsel[r] += acc[r] * scale;
    }
    float* op = ga + ((long)slice * N_ + m0) * N_ + n0 + l16;
#pragma unroll
    for (int r = 0; r < 8; r++) {
        float v = accsel[r];
        if (v == 0.0f) v = -1.0e9f;
        op[(long)(r + 8 * hf) * N_] = v;
    }
}

// ---------------------------------------------------------------- row softmax (in-place, + bf16 copy)
__global__ void softmax_rows(float* __restrict__ ga, bf16_t* __restrict__ abf) {
    __shared__ float red[8];
    __shared__ float sb0, sb1;
    long row = blockIdx.x;
    float* p = ga + row * (long)N_;
    int t = threadIdx.x, w = t >> 5, l = t & 31;

    float v[4];
    float lmax = -3.0e38f;
#pragma unroll
    for (int i = 0; i < 4; i++) { v[i] = p[t + 256 * i]; lmax = fmaxf(lmax, v[i]); }
#pragma unroll
    for (int o = 16; o > 0; o >>= 1) lmax = fmaxf(lmax, __shfl_down(lmax, o));
    if (l == 0) red[w] = lmax;
    __syncthreads();
    if (t == 0) {
        float m = red[0];
        for (int i = 1; i < 8; i++) m = fmaxf(m, red[i]);
        sb0 = m;
    }
    __syncthreads();
    float m = sb0;
    float lsum = 0.0f;
#pragma unroll
    for (int i = 0; i < 4; i++) { v[i] = __expf(v[i] - m); lsum += v[i]; }
#pragma unroll
    for (int o = 16; o > 0; o >>= 1) lsum += __shfl_down(lsum, o);
    __syncthreads();
    if (l == 0) red[w] = lsum;
    __syncthreads();
    if (t == 0) {
        float s = 0.0f;
        for (int i = 0; i < 8; i++) s += red[i];
        sb1 = 1.0f / s;
    }
    __syncthreads();
    float inv = sb1;
#pragma unroll
    for (int i = 0; i < 4; i++) {
        float a = v[i] * inv;
        p[t + 256 * i] = a;
        abf[row * (long)N_ + t + 256 * i] = f2bf(a);
    }
}

// ---------------------------------------------------------------- H-GEMM, stores H transposed (D x N) bf16
__global__ void gemm_h(const bf16_t* __restrict__ Xbf, const bf16_t* __restrict__ Wt,
                       bf16_t* __restrict__ Ht) {
    int slice = blockIdx.y;                                  // h*B + b
    int wave = blockIdx.x * (blockDim.x >> 5) + (threadIdx.x >> 5);
    if (wave >= 64 * 12) return;                             // (1024/16) x (768/64)
    int mt = wave & 63, ng = wave >> 6;
    int lane = threadIdx.x & 31, l16 = lane & 15, hf = lane >> 4;

    const bf16_t* Arow  = Xbf + (long)slice * N_ * D_ + (long)(mt * 16 + l16) * D_;
    const bf16_t* Brow0 = Wt + (long)(ng * 64 + l16) * D_;
    v8f acc[4];
#pragma unroll
    for (int j = 0; j < 4; j++) acc[j] = vzero8();

    v16bf a0 = load_frag_a(Arow, 0, hf);
    v16bf b0[4];
#pragma unroll
    for (int j = 0; j < 4; j++) b0[j] = load_frag_b(Brow0 + (long)(16 * j) * D_, 0, hf);

    for (int k = 32; k < D_; k += 32) {
        __builtin_prefetch(Arow + k + 128, 0, 3);
        v16bf a1 = load_frag_a(Arow, k, hf);
        v16bf b1[4];
#pragma unroll
        for (int j = 0; j < 4; j++) b1[j] = load_frag_b(Brow0 + (long)(16 * j) * D_, k, hf);
#pragma unroll
        for (int j = 0; j < 4; j++) acc[j] = WMMA_BF16(a0, b0[j], acc[j]);
        a0 = a1;
#pragma unroll
        for (int j = 0; j < 4; j++) b0[j] = b1[j];
    }
#pragma unroll
    for (int j = 0; j < 4; j++) acc[j] = WMMA_BF16(a0, b0[j], acc[j]);

    bf16_t* Hs = Ht + (long)slice * D_ * N_;
#pragma unroll
    for (int j = 0; j < 4; j++) {
        int d = ng * 64 + 16 * j + l16;
        v8bf st;
#pragma unroll
        for (int r = 0; r < 8; r++) st[r] = f2bf(acc[j][r]);
        *(v8bf*)(Hs + (long)d * N_ + mt * 16 + 8 * hf) = st;   // 8 contiguous bf16
    }
}

// ---------------------------------------------------------------- neighbor aggregation + fused GCN combine
// sum_nei = attn @ H ; X += relu(sum_nei + gcn_b) ; refresh bf16 copy of X
__global__ void gemm_gcn(const bf16_t* __restrict__ attnbf, const bf16_t* __restrict__ Ht,
                         const float* __restrict__ gcnb, float* __restrict__ Xf,
                         bf16_t* __restrict__ Xbf) {
    int slice = blockIdx.y;
    int wave = blockIdx.x * (blockDim.x >> 5) + (threadIdx.x >> 5);
    if (wave >= 64 * 12) return;
    int mt = wave & 63, ng = wave >> 6;
    int lane = threadIdx.x & 31, l16 = lane & 15, hf = lane >> 4;

    const bf16_t* Arow  = attnbf + (long)slice * N_ * N_ + (long)(mt * 16 + l16) * N_;
    const bf16_t* Brow0 = Ht + (long)slice * D_ * N_ + (long)(ng * 64 + l16) * N_;
    v8f acc[4];
#pragma unroll
    for (int j = 0; j < 4; j++) acc[j] = vzero8();

    v16bf a0 = load_frag_a(Arow, 0, hf);
    v16bf b0[4];
#pragma unroll
    for (int j = 0; j < 4; j++) b0[j] = load_frag_b(Brow0 + (long)(16 * j) * N_, 0, hf);

    for (int k = 32; k < N_; k += 32) {
        __builtin_prefetch(Arow + k + 128, 0, 3);
        v16bf a1 = load_frag_a(Arow, k, hf);
        v16bf b1[4];
#pragma unroll
        for (int j = 0; j < 4; j++) b1[j] = load_frag_b(Brow0 + (long)(16 * j) * N_, k, hf);
#pragma unroll
        for (int j = 0; j < 4; j++) acc[j] = WMMA_BF16(a0, b0[j], acc[j]);
        a0 = a1;
#pragma unroll
        for (int j = 0; j < 4; j++) b0[j] = b1[j];
    }
#pragma unroll
    for (int j = 0; j < 4; j++) acc[j] = WMMA_BF16(a0, b0[j], acc[j]);

#pragma unroll
    for (int j = 0; j < 4; j++) {
        int d = ng * 64 + 16 * j + l16;
        float bs = gcnb[d];
#pragma unroll
        for (int r = 0; r < 8; r++) {
            int m = mt * 16 + r + 8 * hf;
            long idx = (long)slice * N_ * D_ + (long)m * D_ + d;
            float v = acc[j][r] + bs;            // degs == 1 after softmax
            v = v > 0.0f ? v : 0.0f;
            float xn = Xf[idx] + v;
            Xf[idx] = xn;
            Xbf[idx] = f2bf(xn);
        }
    }
}

// ---------------------------------------------------------------- final aggregation GEMM
// out[b][n][:] = cat(X heads) @ W_agg + b_agg ; cat row gathered from (h,B,N,D) layout
__global__ void gemm_agg(const bf16_t* __restrict__ Xbf, const bf16_t* __restrict__ WaggT,
                         const float* __restrict__ bagg, float* __restrict__ Out) {
    const int KTOT = NHEAD_ * D_;                            // 3072
    int wave = blockIdx.x * (blockDim.x >> 5) + (threadIdx.x >> 5);
    if (wave >= 512 * 12) return;                            // (8192/16) x (768/64)
    int mt = wave % 512, ng = wave / 512;
    int lane = threadIdx.x & 31, l16 = lane & 15, hf = lane >> 4;

    int m = mt * 16 + l16;
    int bb = m >> 10, nn = m & (N_ - 1);
    long arowbase = ((long)bb * N_ + nn) * D_;
    const bf16_t* Brow0 = WaggT + (long)(ng * 64 + l16) * KTOT;

    v8f acc[4];
#pragma unroll
    for (int j = 0; j < 4; j++) acc[j] = vzero8();

    v16bf a0 = load_frag_a(Xbf + arowbase, 0, hf);
    v16bf b0[4];
#pragma unroll
    for (int j = 0; j < 4; j++) b0[j] = load_frag_b(Brow0 + (long)(16 * j) * KTOT, 0, hf);

    for (int k = 32; k < KTOT; k += 32) {
        int hh = k / D_, kk = k % D_;
        const bf16_t* Arow = Xbf + (long)hh * B_ * N_ * D_ + arowbase + kk;
        v16bf a1 = load_frag_a(Arow, 0, hf);
        v16bf b1[4];
#pragma unroll
        for (int j = 0; j < 4; j++) b1[j] = load_frag_b(Brow0 + (long)(16 * j) * KTOT, k, hf);
#pragma unroll
        for (int j = 0; j < 4; j++) acc[j] = WMMA_BF16(a0, b0[j], acc[j]);
        a0 = a1;
#pragma unroll
        for (int j = 0; j < 4; j++) b0[j] = b1[j];
    }
#pragma unroll
    for (int j = 0; j < 4; j++) acc[j] = WMMA_BF16(a0, b0[j], acc[j]);

#pragma unroll
    for (int j = 0; j < 4; j++) {
        int n = ng * 64 + 16 * j + l16;
        float bs = bagg[n];
#pragma unroll
        for (int r = 0; r < 8; r++) {
            int mm = mt * 16 + r + 8 * hf;
            Out[(long)mm * D_ + n] = acc[j][r] + bs;
        }
    }
}

// ---------------------------------------------------------------- host launch

extern "C" void kernel_launch(void* const* d_in, const int* in_sizes, int n_in,
                              void* d_out, int out_size, void* d_ws, size_t ws_size,
                              hipStream_t stream) {
    (void)in_sizes; (void)n_in; (void)out_size; (void)ws_size;
    const float* nodes = (const float*)d_in[0];
    const int*   adj   = (const int*)d_in[1];
    const float* Wq    = (const float*)d_in[2];
    const float* bq    = (const float*)d_in[3];
    const float* Wk    = (const float*)d_in[4];
    const float* bk    = (const float*)d_in[5];
    const float* gcnW  = (const float*)d_in[6];
    const float* gcnb  = (const float*)d_in[7];
    const float* Wagg  = (const float*)d_in[8];
    const float* bagg  = (const float*)d_in[9];

    float* out = (float*)d_out;                               // (B,N,D)
    float* ga  = out + (long)B_ * N_ * D_;                    // (h,B,N,N)

    // workspace carve-up (all sub-buffers >=1KB aligned)
    char* ws = (char*)d_ws;
    bf16_t* Xbf_base = (bf16_t*)ws;  ws += (size_t)M_TOT * D_ * 2;
    bf16_t* WqT      = (bf16_t*)ws;  ws += (size_t)NE_ * D_ * D_ * 2;
    bf16_t* WkT      = (bf16_t*)ws;  ws += (size_t)NE_ * D_ * D_ * 2;
    bf16_t* gcnWT    = (bf16_t*)ws;  ws += (size_t)ITERS_ * D_ * D_ * 2;
    bf16_t* WaggT    = (bf16_t*)ws;  ws += (size_t)D_ * NHEAD_ * D_ * 2;
    bf16_t* Qbf      = (bf16_t*)ws;  ws += (size_t)NE_ * M_TOT * D_ * 2;
    bf16_t* Kbf      = (bf16_t*)ws;  ws += (size_t)NE_ * M_TOT * D_ * 2;
    bf16_t* attnbf   = (bf16_t*)ws;  ws += (size_t)NHEAD_ * B_ * N_ * N_ * 2;
    float*  Xf       = (float*)ws;   ws += (size_t)NHEAD_ * B_ * N_ * D_ * 4;
    bf16_t* Xbf_h    = (bf16_t*)ws;  ws += (size_t)NHEAD_ * B_ * N_ * D_ * 2;
    bf16_t* Ht       = (bf16_t*)ws;  ws += (size_t)NHEAD_ * B_ * D_ * N_ * 2;

    // 1) bf16 operand prep
    cvt_f32_to_bf16<<<2048, 256, 0, stream>>>(nodes, Xbf_base, (long)M_TOT * D_);
    transpose_to_bf16<<<dim3(512, 1, NE_),    256, 0, stream>>>(Wq,   WqT,   D_, D_);
    transpose_to_bf16<<<dim3(512, 1, NE_),    256, 0, stream>>>(Wk,   WkT,   D_, D_);
    transpose_to_bf16<<<dim3(512, 1, ITERS_), 256, 0, stream>>>(gcnW, gcnWT, D_, D_);
    transpose_to_bf16<<<dim3(512, 1, 1),      256, 0, stream>>>(Wagg, WaggT, NHEAD_ * D_, D_);

    // 2) Q/K projections per edge: (8192x768) x (768x768) + bias -> bf16
    for (int e = 0; e < NE_; e++) {
        gemm_proj<<<768, 256, 0, stream>>>(Xbf_base, WqT + (long)e * D_ * D_, bq + e * D_,
                                           Qbf + (long)e * M_TOT * D_, M_TOT, D_, D_);
        gemm_proj<<<768, 256, 0, stream>>>(Xbf_base, WkT + (long)e * D_ * D_, bk + e * D_,
                                           Kbf + (long)e * M_TOT * D_, M_TOT, D_, D_);
    }

    // 3) edge-masked attention scores -> ga region of d_out
    scores_kernel<<<dim3(512, NHEAD_ * B_), 256, 0, stream>>>(Qbf, Kbf, adj, ga);

    // 4) row softmax in-place (-> ga output) + bf16 copy for GEMMs
    softmax_rows<<<NHEAD_ * B_ * N_, 256, 0, stream>>>(ga, attnbf);

    // 5) X init: broadcast nodes across heads
    init_X<<<4096, 256, 0, stream>>>(nodes, Xf, Xbf_h, (long)M_TOT * D_,
                                     (long)NHEAD_ * M_TOT * D_);

    // 6) GCN iterations
    for (int i = 0; i < ITERS_; i++) {
        gemm_h<<<dim3(96, NHEAD_ * B_), 256, 0, stream>>>(Xbf_h, gcnWT + (long)i * D_ * D_, Ht);
        gemm_gcn<<<dim3(96, NHEAD_ * B_), 256, 0, stream>>>(attnbf, Ht, gcnb + i * D_, Xf, Xbf_h);
    }

    // 7) final aggregation -> out
    gemm_agg<<<768, 256, 0, stream>>>(Xbf_h, WaggT, bagg, out);
}
